// RITS_652835029738
// MI455X (gfx1250) — matrix operand order repaired and checked by reading, not compiled, not run on hardware
//
#include <hip/hip_runtime.h>
#include <hip/hip_bf16.h>

// ---------------------------------------------------------------------------
// RITS recurrent imputation, CDNA5 (gfx1250) persistent-kernel implementation.
// B=256, L=256, D=128, H=256.  One workgroup per 16-row batch tile (16 WGs),
// 8 waves/WG, full L-loop on device, state in LDS, GEMMs via
// v_wmma_f32_16x16x32_f16 with f32 accumulation.
// Round 2: h double-buffered in LDS; gate GEMM + LSTM fused per j-tile so no
// accumulators live across barriers; partial unrolls to stop B-fragment load
// hoisting -> eliminate scratch spills seen in round 1.
// ---------------------------------------------------------------------------

#define RB 256
#define RL 256
#define RD 128
#define RH 256

typedef _Float16 v16h __attribute__((ext_vector_type(16)));
typedef _Float16 v8h  __attribute__((ext_vector_type(8)));
typedef float    v8f  __attribute__((ext_vector_type(8)));

#define WMMA_F16(acc, a, b) \
  (acc) = __builtin_amdgcn_wmma_f32_16x16x32_f16(false, (a), false, (b), (short)0, (acc), false, false)

// --- A-fragment (16xK tile, f16, row-major in LDS, stride ldk halves) -------
// lane<16 : row M=lane,    halves K = k0+0..7  (v0-3) and k0+16..23 (v4-7)
// lane>=16: row M=lane-16, halves K = k0+8..15 (v0-3) and k0+24..31 (v4-7)
__device__ inline v16h frag_a(const _Float16* base, int ldk, int k0, int lane) {
  const int row = lane & 15;
  const int hi  = lane >> 4;
  const _Float16* p = base + row * ldk + k0 + hi * 8;
  v8h lo = *(const v8h*)(p);
  v8h hh = *(const v8h*)(p + 16);
  v16h r;
#pragma unroll
  for (int i = 0; i < 8; ++i) { r[i] = lo[i]; r[i + 8] = hh[i]; }
  return r;
}

// --- B-fragment (Kx16 tile) from weight W stored row-major [N][K] (f16) ----
// B(k,n) = W[n][k]; lane<16 : col N=n0+lane, K=k0+0..7 and k0+16..23;
//                   lane>=16: col N=n0+lane-16, K=k0+8..15 and k0+24..31.
__device__ inline v16h frag_b(const _Float16* W, int K, int n0, int k0, int lane) {
  const int col = lane & 15;
  const int hi  = lane >> 4;
  const _Float16* p = W + (size_t)(n0 + col) * K + k0 + hi * 8;
  v8h lo = *(const v8h*)(p);
  v8h hh = *(const v8h*)(p + 16);
  v16h r;
#pragma unroll
  for (int i = 0; i < 8; ++i) { r[i] = lo[i]; r[i + 8] = hh[i]; }
  return r;
}

__device__ inline float sigf(float x) { return 1.0f / (1.0f + __expf(-x)); }

// f16 weight regions inside workspace (element offsets in halves)
#define O_TDH   0                       // (H=256) x (D=128)   = 32768
#define O_HIST  32768                   // (D=128) x (H=256)   = 32768
#define O_FEAT  65536                   // (D=128) x (D=128)   = 16384 (diag zeroed)
#define O_WC    81920                   // (D=128) x (2D=256)  = 32768
#define O_WIH   114688                  // (4H=1024) x (2D=256)= 262144
#define O_WHH   376832                  // (4H=1024) x (H=256) = 262144
#define O_TOTAL 638976

// ---------------------------------------------------------------------------
// Prep: convert all weights to f16 (feat_W gets zeroed diagonal), zero the
// per-step loss partial accumulators.
// ---------------------------------------------------------------------------
__global__ __launch_bounds__(256) void rits_prep_kernel(
    const float* __restrict__ td_h_W, const float* __restrict__ hist_W,
    const float* __restrict__ feat_W, const float* __restrict__ wc_W,
    const float* __restrict__ W_ih,   const float* __restrict__ W_hh,
    _Float16* __restrict__ wsh, float* __restrict__ num_g, float* __restrict__ den_g)
{
  size_t i = (size_t)blockIdx.x * blockDim.x + threadIdx.x;
  if (i < 2 * RL) {
    if (i < RL) num_g[i] = 0.0f; else den_g[i - RL] = 0.0f;
  }
  if (i >= O_TOTAL) return;
  float v;
  if (i < O_HIST) {                     // td_h_W
    v = td_h_W[i - O_TDH];
  } else if (i < O_FEAT) {              // hist_W
    v = hist_W[i - O_HIST];
  } else if (i < O_WC) {                // feat_W with zero diagonal
    size_t e = i - O_FEAT;
    int n = (int)(e >> 7), k = (int)(e & 127);
    v = (n == k) ? 0.0f : feat_W[e];
  } else if (i < O_WIH) {               // wc_W
    v = wc_W[i - O_WC];
  } else if (i < O_WHH) {               // W_ih
    v = W_ih[i - O_WIH];
  } else {                              // W_hh
    v = W_hh[i - O_WHH];
  }
  wsh[i] = (_Float16)v;
}

// ---------------------------------------------------------------------------
// Main persistent recurrence kernel: grid 16 blocks x 256 threads (8 waves).
// ---------------------------------------------------------------------------
struct alignas(16) RitsLds {
  float    cbuf[16 * RH];        // LSTM cell state, f32            16 KB
  _Float16 hbuf[2][16 * RH];     // h state, double buffered (f16)  16 KB
  _Float16 in16[16 * (2*RD)];    // [c_c | m]  LSTM input A          8 KB
  _Float16 cat16[16 * (2*RD)];   // [gamma_x | m] alpha A            8 KB
  _Float16 d16[16 * RD];         // deltas tile f16                  4 KB
  _Float16 xc16[16 * RD];        // x_c tile f16                     4 KB
  float    lsum[2];              // per-step loss partials
};

__global__ __launch_bounds__(256) void rits_main_kernel(
    const float* __restrict__ X,   const float* __restrict__ Mk,
    const float* __restrict__ Dl,  const float* __restrict__ Me,
    const float* __restrict__ td_h_b, const float* __restrict__ td_x_w,
    const float* __restrict__ td_x_b, const float* __restrict__ hist_b,
    const float* __restrict__ feat_b, const float* __restrict__ wc_b,
    const float* __restrict__ b_ih,   const float* __restrict__ b_hh,
    const _Float16* __restrict__ wsh,
    float* __restrict__ out, float* __restrict__ num_g, float* __restrict__ den_g)
{
  __shared__ RitsLds s;
  const _Float16* Wth   = wsh + O_TDH;
  const _Float16* Whist = wsh + O_HIST;
  const _Float16* Wfeat = wsh + O_FEAT;
  const _Float16* Wwc   = wsh + O_WC;
  const _Float16* Wih   = wsh + O_WIH;
  const _Float16* Whh   = wsh + O_WHH;

  const int b0     = blockIdx.x * 16;       // 16 batch rows per WG
  const int tid    = threadIdx.x;
  const int lane   = tid & 31;
  const int wave   = tid >> 5;              // 0..7
  const int lane15 = lane & 15;
  const int hi     = lane >> 4;             // 0 or 1

  // zero h (both parities), c state
  for (int i = tid; i < 16 * RH; i += 256) {
    s.hbuf[0][i] = (_Float16)0.0f;
    s.hbuf[1][i] = (_Float16)0.0f;
    s.cbuf[i] = 0.0f;
  }
  __syncthreads();

  for (int t = 0; t < RL; ++t) {
    _Float16* hc = s.hbuf[t & 1];           // current h state (read/decayed)
    _Float16* hn = s.hbuf[(t & 1) ^ 1];     // next h state (written in E/F)

    // ---- Phase A: load deltas/masks; gamma_x; fill A-buffers -------------
    for (int e = tid; e < 16 * RD; e += 256) {
      int mr = e >> 7, col = e & 127;
      size_t g = ((size_t)(b0 + mr) * RL + t) * RD + col;
      float dv = Dl[g], mv = Mk[g];
      s.d16[e] = (_Float16)dv;
      float gx = __expf(-fmaxf(dv * td_x_w[col] + td_x_b[col], 0.0f));
      s.cat16[mr * 256 + col]       = (_Float16)gx;
      s.cat16[mr * 256 + 128 + col] = (_Float16)mv;
      s.in16 [mr * 256 + 128 + col] = (_Float16)mv;
    }
    if (tid == 0) { s.lsum[0] = 0.0f; s.lsum[1] = 0.0f; }
    __syncthreads();

    // ---- Phase B: gamma_h = exp(-relu(d@Wthᵀ+b)); h *= gamma_h (in place)
    {
      v16h a0 = frag_a(s.d16, RD,  0, lane);
      v16h a1 = frag_a(s.d16, RD, 32, lane);
      v16h a2 = frag_a(s.d16, RD, 64, lane);
      v16h a3 = frag_a(s.d16, RD, 96, lane);
#pragma unroll 1
      for (int ti = 0; ti < 2; ++ti) {
        int n0 = (wave * 2 + ti) * 16;
        v8f acc = {};
        WMMA_F16(acc, a0, frag_b(Wth, RD, n0,  0, lane));
        WMMA_F16(acc, a1, frag_b(Wth, RD, n0, 32, lane));
        WMMA_F16(acc, a2, frag_b(Wth, RD, n0, 64, lane));
        WMMA_F16(acc, a3, frag_b(Wth, RD, n0, 96, lane));
        int n = n0 + lane15;
        float bb = td_h_b[n];
#pragma unroll
        for (int r = 0; r < 8; ++r) {
          int m = r + hi * 8;
          float gam = __expf(-fmaxf(acc[r] + bb, 0.0f));
          int idx = m * RH + n;
          float hv = (float)hc[idx] * gam;
          hc[idx] = (_Float16)hv;
        }
      }
    }
    __syncthreads();

    // ---- Phase C: x_h = h@hist_Wᵀ+b ; x_c = m*x+(1-m)*x_h ---------------
    float xhr[8];
    {
      int n0 = wave * 16;
      v8f acc = {};
#pragma unroll 2
      for (int kc = 0; kc < 8; ++kc)
        WMMA_F16(acc, frag_a(hc, RH, kc * 32, lane),
                      frag_b(Whist, RH, n0, kc * 32, lane));
      int n = n0 + lane15;
      float bb = hist_b[n];
#pragma unroll
      for (int r = 0; r < 8; ++r) {
        int m = r + hi * 8;
        float xh = acc[r] + bb;
        xhr[r] = xh;
        size_t g = ((size_t)(b0 + m) * RL + t) * RD + n;
        float xv = X[g], mv = Mk[g];
        s.xc16[m * RD + n] = (_Float16)(mv * xv + (1.0f - mv) * xh);
      }
    }
    __syncthreads();

    // ---- Phase D: z_h, alpha, c_h, c_c, imputations, loss partials ------
    {
      int n0 = wave * 16;
      v8f accZ = {}, accA = {};
#pragma unroll 2
      for (int kc = 0; kc < 4; ++kc)
        WMMA_F16(accZ, frag_a(s.xc16, RD, kc * 32, lane),
                       frag_b(Wfeat, RD, n0, kc * 32, lane));
#pragma unroll 2
      for (int kc = 0; kc < 8; ++kc)
        WMMA_F16(accA, frag_a(s.cat16, 2 * RD, kc * 32, lane),
                       frag_b(Wwc, 2 * RD, n0, kc * 32, lane));
      int n = n0 + lane15;
      float fb = feat_b[n], wb = wc_b[n];
      float num = 0.0f, den = 0.0f;
#pragma unroll
      for (int r = 0; r < 8; ++r) {
        int m = r + hi * 8;
        float alpha = accA[r] + wb;
        float zh    = accZ[r] + fb;
        float ch    = alpha * zh + (1.0f - alpha) * xhr[r];
        size_t g = ((size_t)(b0 + m) * RL + t) * RD + n;
        float xv = X[g], mv = Mk[g], mev = Me[g];
        float cc = mv * xv + (1.0f - mv) * ch;
        out[g] = cc;                               // imputations (B,L,D)
        s.in16[m * 256 + n] = (_Float16)cc;        // LSTM input cols 0..127
        num += fabsf(cc - xv) * mev;
        den += mev;
      }
      atomicAdd(&s.lsum[0], num);
      atomicAdd(&s.lsum[1], den);
    }
    __syncthreads();

    // ---- Phase E/F fused: gates (i,f,g,o) per j-tile, LSTM in registers,
    //      write NEXT h buffer (no barrier between GEMM reads and h writes,
    //      accumulators never live across a sync) ------------------------
#pragma unroll 1
    for (int ti = 0; ti < 2; ++ti) {
      const int jt = wave * 2 + ti;               // hidden-unit tile 0..15
      v8f acc0 = {}, acc1 = {}, acc2 = {}, acc3 = {};
#pragma unroll 2
      for (int kc = 0; kc < 8; ++kc) {
        v16h a = frag_a(s.in16, 2 * RD, kc * 32, lane);
        WMMA_F16(acc0, a, frag_b(Wih, 2 * RD, 0 * RH + jt * 16, kc * 32, lane));
        WMMA_F16(acc1, a, frag_b(Wih, 2 * RD, 1 * RH + jt * 16, kc * 32, lane));
        WMMA_F16(acc2, a, frag_b(Wih, 2 * RD, 2 * RH + jt * 16, kc * 32, lane));
        WMMA_F16(acc3, a, frag_b(Wih, 2 * RD, 3 * RH + jt * 16, kc * 32, lane));
      }
#pragma unroll 2
      for (int kc = 0; kc < 8; ++kc) {
        v16h a = frag_a(hc, RH, kc * 32, lane);
        WMMA_F16(acc0, a, frag_b(Whh, RH, 0 * RH + jt * 16, kc * 32, lane));
        WMMA_F16(acc1, a, frag_b(Whh, RH, 1 * RH + jt * 16, kc * 32, lane));
        WMMA_F16(acc2, a, frag_b(Whh, RH, 2 * RH + jt * 16, kc * 32, lane));
        WMMA_F16(acc3, a, frag_b(Whh, RH, 3 * RH + jt * 16, kc * 32, lane));
      }
      const int j = jt * 16 + lane15;
      float bi = b_ih[j]          + b_hh[j];
      float bf = b_ih[RH + j]     + b_hh[RH + j];
      float bg = b_ih[2 * RH + j] + b_hh[2 * RH + j];
      float bo = b_ih[3 * RH + j] + b_hh[3 * RH + j];
#pragma unroll
      for (int r = 0; r < 8; ++r) {
        int m = r + hi * 8;
        float ig = sigf (acc0[r] + bi);
        float fg = sigf (acc1[r] + bf);
        float gv = tanhf(acc2[r] + bg);
        float og = sigf (acc3[r] + bo);
        int idx = m * RH + j;
        float c = fg * s.cbuf[idx] + ig * gv;
        s.cbuf[idx] = c;
        hn[idx] = (_Float16)(og * tanhf(c));      // next-step h
      }
    }
    if (tid == 0) {
      atomicAdd(&num_g[t], s.lsum[0]);
      atomicAdd(&den_g[t], s.lsum[1]);
    }
    __syncthreads();
  }
}

// ---------------------------------------------------------------------------
// Final loss reduction: loss = sum_t num[t] / (den[t] + 1e-5)
// ---------------------------------------------------------------------------
__global__ __launch_bounds__(256) void rits_loss_kernel(
    const float* __restrict__ num_g, const float* __restrict__ den_g,
    float* __restrict__ out_loss)
{
  __shared__ float red[256];
  int t = threadIdx.x;
  red[t] = num_g[t] / (den_g[t] + 1e-5f);
  __syncthreads();
  for (int ss = 128; ss > 0; ss >>= 1) {
    if (t < ss) red[t] += red[t + ss];
    __syncthreads();
  }
  if (t == 0) *out_loss = red[0];   // IMPUTE_WEIGHT == 1.0
}

// ---------------------------------------------------------------------------
extern "C" void kernel_launch(void* const* d_in, const int* in_sizes, int n_in,
                              void* d_out, int out_size, void* d_ws, size_t ws_size,
                              hipStream_t stream) {
  const float* X      = (const float*)d_in[0];   // values   (B,L,D)
  const float* Mk     = (const float*)d_in[1];   // masks
  const float* Dl     = (const float*)d_in[2];   // deltas
  const float* Me     = (const float*)d_in[3];   // eval_masks
  const float* td_h_W = (const float*)d_in[4];   // (H,D)
  const float* td_h_b = (const float*)d_in[5];
  const float* td_x_w = (const float*)d_in[6];
  const float* td_x_b = (const float*)d_in[7];
  const float* hist_W = (const float*)d_in[8];   // (D,H)
  const float* hist_b = (const float*)d_in[9];
  const float* feat_W = (const float*)d_in[10];  // (D,D)
  const float* feat_b = (const float*)d_in[11];
  const float* wc_W   = (const float*)d_in[12];  // (D,2D)
  const float* wc_b   = (const float*)d_in[13];
  const float* W_ih   = (const float*)d_in[14];  // (4H,2D)
  const float* W_hh   = (const float*)d_in[15];  // (4H,H)
  const float* b_ih   = (const float*)d_in[16];
  const float* b_hh   = (const float*)d_in[17];

  _Float16* wsh   = (_Float16*)d_ws;                                  // f16 weights
  float*    num_g = (float*)((char*)d_ws + (size_t)O_TOTAL * sizeof(_Float16));
  float*    den_g = num_g + RL;

  float* out      = (float*)d_out;                                    // (B,L,D)
  float* out_loss = out + (size_t)RB * RL * RD;

  // 1) convert weights + zero loss accumulators
  int nprep = (O_TOTAL + 255) / 256;
  rits_prep_kernel<<<nprep, 256, 0, stream>>>(td_h_W, hist_W, feat_W, wc_W,
                                              W_ih, W_hh, wsh, num_g, den_g);
  // 2) persistent recurrence: 16 batch tiles x 8 waves
  rits_main_kernel<<<RB / 16, 256, 0, stream>>>(X, Mk, Dl, Me,
                                                td_h_b, td_x_w, td_x_b,
                                                hist_b, feat_b, wc_b,
                                                b_ih, b_hh, wsh,
                                                out, num_g, den_g);
  // 3) loss reduction
  rits_loss_kernel<<<1, 256, 0, stream>>>(num_g, den_g, out_loss);
}
